// GAT_GraphEncoder_15942918603108
// MI455X (gfx1250) — matrix operand order repaired and checked by reading, not compiled, not run on hardware
//
#include <hip/hip_runtime.h>
#include <hip/hip_bf16.h>

// ---------------- problem constants ----------------
#define BB   16
#define NN   512
#define DD   256
#define HH   8
#define DHD  32
#define DFF  1024
#define LL   2
#define NEGINF (-1e9f)

typedef __attribute__((ext_vector_type(16))) __bf16 v16bf;
typedef __attribute__((ext_vector_type(8)))  float  v8f;

union FragBF { v16bf v; unsigned int u[8]; };

static __device__ __forceinline__ unsigned short f2bf(float f) {
  union { float f; unsigned int u; } c; c.f = f;
  unsigned int r = c.u + 0x7FFFu + ((c.u >> 16) & 1u);   // round-to-nearest-even
  return (unsigned short)(r >> 16);
}

// ---------------- block reductions (256 threads) ----------------
static __device__ __forceinline__ float blkRedSum(float v, float* s) {
  int tid = threadIdx.x;
  s[tid] = v; __syncthreads();
  for (int off = 128; off > 0; off >>= 1) {
    if (tid < off) s[tid] += s[tid + off];
    __syncthreads();
  }
  float r = s[0]; __syncthreads();
  return r;
}
static __device__ __forceinline__ float blkRedMax(float v, float* s) {
  int tid = threadIdx.x;
  s[tid] = v; __syncthreads();
  for (int off = 128; off > 0; off >>= 1) {
    if (tid < off) s[tid] = fmaxf(s[tid], s[tid + off]);
    __syncthreads();
  }
  float r = s[0]; __syncthreads();
  return r;
}

// ---------------- generic batched GEMM: C = act(A @ B(^T) + bias) ----------------
// A: [batch] M x K row-major (f32), B: [batch] K x N (or N x K if transB), C: [batch] M x N
// f32 -> bf16 conversion while staging tiles to LDS; WMMA bf16 with f32 accumulation.
// Tile 128x128x32, 256 threads (8 waves, 4x2), wave computes 32x64 = 2x4 WMMA frags.
__global__ __launch_bounds__(256)
void k_gemm(const float* __restrict__ A, const float* __restrict__ B,
            const float* __restrict__ bias, float* __restrict__ C,
            int M, int N, int K,
            long long sA, long long sB, long long sC,
            int transB, int act)
{
  __shared__ unsigned short As[128][40];   // [m][k], padded to kill bank conflicts
  __shared__ unsigned short Bs[128][40];   // [n][k]

  const int tid  = threadIdx.x;
  const int Mblk = blockIdx.x * 128;
  const int Nblk = blockIdx.y * 128;
  A += (long long)blockIdx.z * sA;
  B += (long long)blockIdx.z * sB;
  C += (long long)blockIdx.z * sC;

  v8f acc[2][4];
  #pragma unroll
  for (int i = 0; i < 2; ++i)
    #pragma unroll
    for (int j = 0; j < 4; ++j)
      #pragma unroll
      for (int r = 0; r < 8; ++r) acc[i][j][r] = 0.0f;

  const int lane = tid & 31, wid = tid >> 5;
  const int wm = (wid >> 1) * 32;      // 0,32,64,96
  const int wn = (wid & 1) * 64;       // 0,64
  const int half = lane >> 4, l16 = lane & 15;

  for (int k0 = 0; k0 < K; k0 += 32) {
    // ---- stage A tile: 128 rows x 32 k, float4 loads, bf16 packed stores ----
    #pragma unroll
    for (int t = 0; t < 4; ++t) {
      int idx = tid + t * 256;
      int row = idx >> 3;
      int kq  = (idx & 7) * 4;
      float4 f = make_float4(0.f, 0.f, 0.f, 0.f);
      if (Mblk + row < M)
        f = *(const float4*)(A + (long long)(Mblk + row) * K + k0 + kq);
      unsigned int p0 = (unsigned int)f2bf(f.x) | ((unsigned int)f2bf(f.y) << 16);
      unsigned int p1 = (unsigned int)f2bf(f.z) | ((unsigned int)f2bf(f.w) << 16);
      *(unsigned int*)&As[row][kq]     = p0;
      *(unsigned int*)&As[row][kq + 2] = p1;
    }
    // ---- stage B tile as Bs[n][k] ----
    if (transB) {          // B is N x K row-major: same access pattern as A
      #pragma unroll
      for (int t = 0; t < 4; ++t) {
        int idx = tid + t * 256;
        int n  = idx >> 3;
        int kq = (idx & 7) * 4;
        float4 f = make_float4(0.f, 0.f, 0.f, 0.f);
        if (Nblk + n < N)
          f = *(const float4*)(B + (long long)(Nblk + n) * K + k0 + kq);
        unsigned int p0 = (unsigned int)f2bf(f.x) | ((unsigned int)f2bf(f.y) << 16);
        unsigned int p1 = (unsigned int)f2bf(f.z) | ((unsigned int)f2bf(f.w) << 16);
        *(unsigned int*)&Bs[n][kq]     = p0;
        *(unsigned int*)&Bs[n][kq + 2] = p1;
      }
    } else {               // B is K x N row-major: load along n, transpose into LDS
      #pragma unroll
      for (int t = 0; t < 4; ++t) {
        int idx = tid + t * 256;
        int kr = idx >> 5;           // 0..31
        int nq = (idx & 31) * 4;     // 0..124
        float4 f = make_float4(0.f, 0.f, 0.f, 0.f);
        if (Nblk + nq + 3 < N)
          f = *(const float4*)(B + (long long)(k0 + kr) * N + Nblk + nq);
        Bs[nq + 0][kr] = f2bf(f.x);
        Bs[nq + 1][kr] = f2bf(f.y);
        Bs[nq + 2][kr] = f2bf(f.z);
        Bs[nq + 3][kr] = f2bf(f.w);
      }
    }
    if (k0 + 32 < K)
      __builtin_prefetch(A + (long long)(Mblk + (tid >> 1)) * K + k0 + 32, 0, 1);
    __syncthreads();

    // ---- build fragments per ISA 16-bit layouts and issue WMMAs ----
    FragBF fa[2], fb[4];
    #pragma unroll
    for (int i = 0; i < 2; ++i) {
      int row = wm + i * 16 + l16;
      #pragma unroll
      for (int v = 0; v < 8; ++v) {
        int kk = ((v < 4) ? (2 * v) : (16 + 2 * (v - 4))) + 8 * half;  // A layout
        fa[i].u[v] = *(const unsigned int*)&As[row][kk];
      }
    }
    #pragma unroll
    for (int j = 0; j < 4; ++j) {
      int col = wn + j * 16 + l16;
      #pragma unroll
      for (int v = 0; v < 8; ++v) {
        int kk = ((v < 4) ? (2 * v) : (16 + 2 * (v - 4))) + 8 * half;  // B layout
        fb[j].u[v] = *(const unsigned int*)&Bs[col][kk];
      }
    }
    #pragma unroll
    for (int i = 0; i < 2; ++i)
      #pragma unroll
      for (int j = 0; j < 4; ++j)
        acc[i][j] = __builtin_amdgcn_wmma_f32_16x16x32_bf16(
            false, fa[i].v, false, fb[j].v, (short)0, acc[i][j], false, false);
    __syncthreads();
  }

  // ---- epilogue: bias + activation, guarded store ----
  #pragma unroll
  for (int i = 0; i < 2; ++i) {
    int mbase = Mblk + wm + i * 16 + 8 * half;
    #pragma unroll
    for (int j = 0; j < 4; ++j) {
      int n = Nblk + wn + j * 16 + l16;
      if (n < N) {
        float bv = bias ? bias[n] : 0.0f;
        #pragma unroll
        for (int r = 0; r < 8; ++r) {
          int m = mbase + r;
          if (m < M) {
            float val = acc[i][j][r] + bv;
            if (act == 1) val = fmaxf(val, 0.0f);
            C[(long long)m * N + n] = val;
          }
        }
      }
    }
  }
}

// ---------------- embedding gather ----------------
__global__ __launch_bounds__(256)
void k_embed(const int* __restrict__ idx, const float* __restrict__ emb,
             float* __restrict__ x)
{
  int r = blockIdx.x, d = threadIdx.x;
  x[(long long)r * DD + d] = emb[(long long)idx[r] * DD + d];
}

// ---------------- GAT: s1 = h@a1, s2 = h@a2 (per-row dots) ----------------
__global__ __launch_bounds__(256)
void k_gat_scores(const float* __restrict__ h, const float* __restrict__ a1,
                  const float* __restrict__ a2, float* __restrict__ s1,
                  float* __restrict__ s2)
{
  __shared__ float sbuf[256];
  int r = blockIdx.x, d = threadIdx.x;
  float hv = h[(long long)r * DD + d];
  float v1 = blkRedSum(hv * a1[d], sbuf);
  float v2 = blkRedSum(hv * a2[d], sbuf);
  if (d == 0) { s1[r] = v1; s2[r] = v2; }
}

// ---------------- GAT masked edge softmax: alpha[b,i,:] ----------------
__global__ __launch_bounds__(256)
void k_gat_softmax(const float* __restrict__ s1, const float* __restrict__ s2,
                   const int* __restrict__ adj, float* __restrict__ alpha)
{
  __shared__ float sbuf[256];
  int r = blockIdx.x;            // r = b*NN + i
  int b = r >> 9;
  long long base = (long long)r * NN;
  const int* adjrow = adj + base;
  float si = s1[r];
  float e[2];
  #pragma unroll
  for (int p = 0; p < 2; ++p) {
    int j = threadIdx.x + p * 256;
    float eij = si + s2[b * NN + j];
    eij = (eij > 0.0f) ? eij : 0.2f * eij;       // leaky relu
    e[p] = (adjrow[j] > 0) ? eij : NEGINF;
  }
  float mx = blkRedMax(fmaxf(e[0], e[1]), sbuf);
  float ex0 = __expf(e[0] - mx), ex1 = __expf(e[1] - mx);
  float sum = blkRedSum(ex0 + ex1, sbuf);
  float inv = 1.0f / sum;
  alpha[base + threadIdx.x]       = ex0 * inv;
  alpha[base + threadIdx.x + 256] = ex1 * inv;
}

// ---------------- ELU ----------------
__global__ __launch_bounds__(256)
void k_elu(const float* __restrict__ in, float* __restrict__ out)
{
  long long t = (long long)blockIdx.x * 256 + threadIdx.x;
  float v = in[t];
  out[t] = (v > 0.0f) ? v : (__expf(v) - 1.0f);
}

// ---------------- x = x*sqrt(D) + PE(pos,d) ----------------
__global__ __launch_bounds__(256)
void k_scale_pe(float* __restrict__ x)
{
  long long t = (long long)blockIdx.x * 256 + threadIdx.x;
  int d = (int)(t & (DD - 1));
  int pos = (int)((t >> 8) & (NN - 1));
  float i2 = (float)((d >> 1) * 2);
  float ang = (float)pos * __powf(10000.0f, -i2 / (float)DD);
  float pe = (d & 1) ? __cosf(ang) : __sinf(ang);
  x[t] = x[t] * 16.0f + pe;      // sqrt(256) = 16
}

// ---------------- [B,N,H,dh] -> [B,H,N,dh] ----------------
__global__ __launch_bounds__(256)
void k_to_heads(const float* __restrict__ in, float* __restrict__ out)
{
  long long t = (long long)blockIdx.x * 256 + threadIdx.x;   // out index
  int e = (int)(t & 31);
  int n = (int)((t >> 5) & (NN - 1));
  int h = (int)((t >> 14) & (HH - 1));
  int b = (int)(t >> 17);
  out[t] = in[((((long long)b * NN + n) * HH + h) << 5) + e];
}

// ---------------- [B,H,N,dh] -> [B,N,H,dh] ----------------
__global__ __launch_bounds__(256)
void k_from_heads(const float* __restrict__ in, float* __restrict__ out)
{
  long long t = (long long)blockIdx.x * 256 + threadIdx.x;   // out index
  int e = (int)(t & 31);
  int h = (int)((t >> 5) & (HH - 1));
  int n = (int)((t >> 8) & (NN - 1));
  int b = (int)(t >> 17);
  out[t] = in[((((long long)b * HH + h) * NN + n) << 5) + e];
}

// ---------------- attention softmax (in place, scaled + padding mask) ----------------
__global__ __launch_bounds__(256)
void k_attn_softmax(float* __restrict__ sc, const int* __restrict__ mask)
{
  __shared__ float sbuf[256];
  const float scale = 0.17677669529663687f;    // 1/sqrt(32)
  long long row = blockIdx.x;                  // row = ((b*H + h)*N + q)
  int b = (int)(row >> 12);                    // H*N = 4096
  long long base = row * NN;
  float e[2];
  #pragma unroll
  for (int p = 0; p < 2; ++p) {
    int j = threadIdx.x + p * 256;
    e[p] = sc[base + j] * scale + (float)mask[b * NN + j] * NEGINF;
  }
  float mx = blkRedMax(fmaxf(e[0], e[1]), sbuf);
  float ex0 = __expf(e[0] - mx), ex1 = __expf(e[1] - mx);
  float sum = blkRedSum(ex0 + ex1, sbuf);
  float inv = 1.0f / sum;
  sc[base + threadIdx.x]       = ex0 * inv;
  sc[base + threadIdx.x + 256] = ex1 * inv;
}

// ---------------- out = LayerNorm(res + t) * g + beta ----------------
__global__ __launch_bounds__(256)
void k_ln(const float* __restrict__ res, const float* __restrict__ tin,
          const float* __restrict__ g, const float* __restrict__ beta,
          float* __restrict__ out)
{
  __shared__ float sbuf[256];
  int r = blockIdx.x, d = threadIdx.x;
  long long i = (long long)r * DD + d;
  float v = res[i] + tin[i];
  float mu = blkRedSum(v, sbuf) * (1.0f / DD);
  float c = v - mu;
  float var = blkRedSum(c * c, sbuf) * (1.0f / DD);
  out[i] = c * rsqrtf(var + 1e-6f) * g[d] + beta[d];
}

// ---------------- host orchestration ----------------
static void gemm(const float* A, const float* B, const float* bias, float* C,
                 int M, int N, int K, long long sA, long long sB, long long sC,
                 int transB, int act, int batch, hipStream_t s)
{
  dim3 grid((M + 127) / 128, (N + 127) / 128, batch);
  k_gemm<<<grid, dim3(256), 0, s>>>(A, B, bias, C, M, N, K, sA, sB, sC, transB, act);
}

extern "C" void kernel_launch(void* const* d_in, const int* in_sizes, int n_in,
                              void* d_out, int out_size, void* d_ws, size_t ws_size,
                              hipStream_t stream)
{
  (void)in_sizes; (void)n_in; (void)out_size; (void)ws_size;
  const int*   node = (const int*)d_in[0];
  const int*   edge = (const int*)d_in[1];
  const int*   mask = (const int*)d_in[2];
  const float* emb  = (const float*)d_in[4];
  const float* Wg   = (const float*)d_in[5];
  const float* a1   = (const float*)d_in[6];
  const float* a2   = (const float*)d_in[7];
  const float* Wq   = (const float*)d_in[8];
  const float* bq   = (const float*)d_in[9];
  const float* Wk   = (const float*)d_in[10];
  const float* bk   = (const float*)d_in[11];
  const float* Wv   = (const float*)d_in[12];
  const float* bv   = (const float*)d_in[13];
  const float* Wo   = (const float*)d_in[14];
  const float* bo   = (const float*)d_in[15];
  const float* W1   = (const float*)d_in[16];
  const float* b1   = (const float*)d_in[17];
  const float* W2   = (const float*)d_in[18];
  const float* b2   = (const float*)d_in[19];
  const float* ln1g = (const float*)d_in[20];
  const float* ln1b = (const float*)d_in[21];
  const float* ln2g = (const float*)d_in[22];
  const float* ln2b = (const float*)d_in[23];

  const long long XE = (long long)BB * NN * DD;        // 2,097,152
  float* ws = (float*)d_ws;
  float* x     = ws;
  float* h     = x + XE;
  float* t0    = h + XE;                               // [8192,1024] FFN mid
  float* q     = t0 + (long long)BB * NN * DFF;
  float* kbuf  = q + XE;
  float* vbuf  = kbuf + XE;
  float* t1    = vbuf + XE;
  float* o     = t1 + XE;
  float* alpha = o + XE;                               // [B,N,N]
  float* s1    = alpha + (long long)BB * NN * NN;
  float* s2    = s1 + BB * NN;

  float* outx    = (float*)d_out;                      // [B,N,D]
  float* attnbuf = outx + XE;                          // [B,H,N,N] (output #2)

  const int ROWS = BB * NN;                            // 8192
  dim3 blk(256);

  // embedding lookup
  k_embed<<<ROWS, blk, 0, stream>>>(node, emb, x);

  // ---- GAT hops (shared weights) ----
  for (int hop = 0; hop < 2; ++hop) {
    gemm(x, Wg, nullptr, h, ROWS, DD, DD, 0, 0, 0, 0, 0, 1, stream);
    k_gat_scores<<<ROWS, blk, 0, stream>>>(h, a1, a2, s1, s2);
    k_gat_softmax<<<ROWS, blk, 0, stream>>>(s1, s2, edge, alpha);
    gemm(alpha, h, nullptr, t1, NN, DD, NN,
         (long long)NN * NN, (long long)NN * DD, (long long)NN * DD, 0, 0, BB, stream);
    k_elu<<<ROWS, blk, 0, stream>>>(t1, x);
  }

  // ---- scale + positional encoding ----
  k_scale_pe<<<ROWS, blk, 0, stream>>>(x);

  // ---- transformer encoder stack ----
  const long long hs = (long long)NN * DHD;            // per-(b,h) Q/K/V stride
  const long long ss = (long long)NN * NN;             // per-(b,h) scores stride
  for (int l = 0; l < LL; ++l) {
    const float* Wq_l = Wq + (long long)l * DD * DD;
    const float* Wk_l = Wk + (long long)l * DD * DD;
    const float* Wv_l = Wv + (long long)l * DD * DD;
    const float* Wo_l = Wo + (long long)l * DD * DD;
    const float* W1_l = W1 + (long long)l * DD * DFF;
    const float* W2_l = W2 + (long long)l * DFF * DD;

    gemm(x, Wq_l, bq + l * DD, t1, ROWS, DD, DD, 0, 0, 0, 0, 0, 1, stream);
    k_to_heads<<<ROWS, blk, 0, stream>>>(t1, q);
    gemm(x, Wk_l, bk + l * DD, t1, ROWS, DD, DD, 0, 0, 0, 0, 0, 1, stream);
    k_to_heads<<<ROWS, blk, 0, stream>>>(t1, kbuf);
    gemm(x, Wv_l, bv + l * DD, t1, ROWS, DD, DD, 0, 0, 0, 0, 0, 1, stream);
    k_to_heads<<<ROWS, blk, 0, stream>>>(t1, vbuf);

    // scores = Q @ K^T  (batched over B*H, directly into d_out attn region)
    gemm(q, kbuf, nullptr, attnbuf, NN, NN, DHD, hs, hs, ss, 1, 0, BB * HH, stream);
    k_attn_softmax<<<BB * HH * NN, blk, 0, stream>>>(attnbuf, mask);
    // O = attn @ V
    gemm(attnbuf, vbuf, nullptr, o, NN, DHD, NN, ss, hs, hs, 0, 0, BB * HH, stream);
    k_from_heads<<<ROWS, blk, 0, stream>>>(o, t1);
    gemm(t1, Wo_l, bo + l * DD, h, ROWS, DD, DD, 0, 0, 0, 0, 0, 1, stream);
    k_ln<<<ROWS, blk, 0, stream>>>(x, h, ln1g + l * DD, ln1b + l * DD, x);

    gemm(x, W1_l, b1 + l * DFF, t0, ROWS, DFF, DD, 0, 0, 0, 0, 1, 1, stream);
    gemm(t0, W2_l, b2 + l * DD, h, ROWS, DD, DFF, 0, 0, 0, 0, 0, 1, stream);
    float* dst = (l == LL - 1) ? outx : x;
    k_ln<<<ROWS, blk, 0, stream>>>(x, h, ln2g + l * DD, ln2b + l * DD, dst);
  }
}
